// Transformer_59957743452690
// MI455X (gfx1250) — matrix operand order repaired and checked by reading, not compiled, not run on hardware
//
#include <hip/hip_runtime.h>
#include <hip/hip_bf16.h>

typedef __bf16 bf16;
typedef __attribute__((ext_vector_type(16))) __bf16 v16bf;
typedef __attribute__((ext_vector_type(8)))  __bf16 bf16x8;
typedef __attribute__((ext_vector_type(8)))  float  v8f;
typedef __attribute__((ext_vector_type(4)))  unsigned int u32x4;
typedef __attribute__((ext_vector_type(8)))  int i32x8;
typedef __attribute__((ext_vector_type(4)))  int i32x4;
typedef __attribute__((address_space(1))) void gvoid;
typedef __attribute__((address_space(3))) void lvoid;

#define WMMA_BF16(a, b, c) \
    __builtin_amdgcn_wmma_f32_16x16x32_bf16(false, (a), false, (b), (short)0, (c), false, false)

// --- CDNA5 data-mover path selection (all probe-guarded) ---------------------
#if __has_builtin(__builtin_amdgcn_tensor_load_to_lds) && \
    __has_builtin(__builtin_amdgcn_s_wait_tensorcnt)
#define USE_TDM 1
#elif __has_builtin(__builtin_amdgcn_global_load_async_to_lds_b128)
#define USE_ASYNC 1
#endif

static __device__ __forceinline__ v16bf join8(bf16x8 lo, bf16x8 hi) {
    return __builtin_shufflevector(lo, hi, 0,1,2,3,4,5,6,7,8,9,10,11,12,13,14,15);
}

#if defined(USE_TDM)
// Issue one TDM descriptor moving a 128-row x 32-col bf16 tile (row-major,
// row stride = strideElems) from global to LDS. D# layout per ISA ch.8.
// 6-arg toolchain variant: (g0, g1, g2, g3, g_extra, cpol).
static __device__ __forceinline__ void tdm_load_tile_128x32(
        const bf16* gsrc, void* ldst, int strideElems, int rowsTotal, int colsTotal) {
    const unsigned long long ga = (unsigned long long)(uintptr_t)gsrc;
    const unsigned lds = (unsigned)(uintptr_t)ldst;  // low 32 bits = LDS byte offset
    u32x4 g0;
    g0[0] = 1u;                                               // count=1, user descriptor
    g0[1] = lds;                                              // lds_addr
    g0[2] = (unsigned)ga;                                     // global_addr[31:0]
    g0[3] = (unsigned)((ga >> 32) & 0x01FFFFFFu) | (2u << 30); // addr[56:32] | type=2
    i32x8 g1;
    g1[0] = (int)(1u << 16);                                  // data_size = 2 bytes
    g1[1] = (int)(((unsigned)colsTotal & 0xFFFFu) << 16);     // tensor_dim0[15:0]
    g1[2] = (int)((((unsigned)colsTotal >> 16) & 0xFFFFu) |
                  (((unsigned)rowsTotal & 0xFFFFu) << 16));   // dim0 hi | dim1 lo
    g1[3] = (int)((((unsigned)rowsTotal >> 16) & 0xFFFFu) |
                  (32u << 16));                               // dim1 hi | tile_dim0=32
    g1[4] = (int)128u;                                        // tile_dim1=128, tile_dim2=0
    g1[5] = (int)(unsigned)strideElems;                       // tensor_dim0_stride lo
    g1[6] = 0;                                                // stride hi | dim1_stride lo
    g1[7] = 0;
    i32x4 z4 = {};
    i32x8 z8 = {};
    __builtin_amdgcn_tensor_load_to_lds(g0, g1, z4, z4, z8, 0);
}
#endif

#if defined(USE_ASYNC)
static __device__ __forceinline__ void async_b128(const bf16* gsrc, bf16* ldst) {
    __builtin_amdgcn_global_load_async_to_lds_b128(
        (gvoid*)(uintptr_t)gsrc, (lvoid*)(uintptr_t)ldst, 0, 0);
}
static __device__ __forceinline__ void wait_async0() {
#if __has_builtin(__builtin_amdgcn_s_wait_asynccnt)
    __builtin_amdgcn_s_wait_asynccnt(0);
#else
    asm volatile("s_wait_asynccnt 0" ::: "memory");
#endif
}
#endif

// ---------------------------------------------------------------------------
// Elementwise f32 -> bf16 conversion (weights)
// ---------------------------------------------------------------------------
__global__ void cvt_f32_bf16(const float* __restrict__ s, bf16* __restrict__ d, int n) {
    for (int i = blockIdx.x * blockDim.x + threadIdx.x; i < n; i += gridDim.x * blockDim.x)
        d[i] = (bf16)s[i];
}

// ---------------------------------------------------------------------------
// RMSNorm: one block per row of D elements, f32 in, bf16 out
// ---------------------------------------------------------------------------
__global__ __launch_bounds__(256)
void rmsnorm_kernel(const float* __restrict__ x, const float* __restrict__ g,
                    bf16* __restrict__ out, int D) {
    __shared__ float red[256];
    const int tid = threadIdx.x;
    const size_t row = blockIdx.x;
    const float* xr = x + row * D;
    float ss = 0.f;
    for (int i = tid; i < D; i += 256) { float v = xr[i]; ss += v * v; }
    red[tid] = ss;
    __syncthreads();
    for (int s = 128; s > 0; s >>= 1) {
        if (tid < s) red[tid] += red[tid + s];
        __syncthreads();
    }
    const float inv = rsqrtf(red[0] / (float)D + 1e-5f);
    for (int i = tid; i < D; i += 256)
        out[row * D + i] = (bf16)(xr[i] * inv * g[i]);
}

// ---------------------------------------------------------------------------
// Tiled WMMA bf16 GEMM:  Y[M,N] = A[M,K] @ W[N,K]^T
// Block tile 128x128x32, 8 waves (2 along M x 4 along N), wave tile 64x32.
// Double-buffered LDS; tiles staged by TDM DMA (preferred), ASYNC global->LDS,
// or VGPR fallback. Prefetch of stage s+1 overlaps the 8 WMMAs of stage s.
// EPI: 0 = store bf16
//      1 = gelu(erf) then store bf16
//      2 = f32 out = val + residual
//      3 = f32 out += val (accumulate in place)
//      4 = store bf16 transposed per batch: OB[b][n][s]  (V^T layout)
// ---------------------------------------------------------------------------
template <int EPI>
__global__ __launch_bounds__(256)
void gemm_bf16_wmma(const bf16* __restrict__ A, const bf16* __restrict__ W,
                    const float* __restrict__ R, float* __restrict__ OF,
                    bf16* __restrict__ OB, int M, int N, int K, int Sdim) {
    __shared__ bf16 sA[2][128 * 32];
    __shared__ bf16 sW[2][128 * 32];

    const int tid  = threadIdx.x;
    const int lane = tid & 31;
    const int wave = tid >> 5;
    const int r16  = lane & 15;
    const int hi   = lane >> 4;           // 0: lanes 0-15, 1: lanes 16-31
    const int akb  = hi * 16;             // A-frag k base (contiguous k, matches B)
    const int bkb  = hi * 16;             // B-frag k base (contiguous k, matches A)
    const int wm   = (wave >> 2) * 64;    // wave M offset in block tile
    const int wn   = (wave & 3) * 32;     // wave N offset in block tile

    const int m0 = blockIdx.y * 128;
    const int n0 = blockIdx.x * 128;

    v8f acc[4][2] = {};

    auto issue = [&](int st, int k0) {
#if defined(USE_TDM)
        if (wave == 0) {
            tdm_load_tile_128x32(A + (size_t)m0 * K + k0, &sA[st][0], K, M, K);
            tdm_load_tile_128x32(W + (size_t)n0 * K + k0, &sW[st][0], K, N, K);
        }
#elif defined(USE_ASYNC)
        const int row = tid >> 1, h2 = (tid & 1) * 16;
        const bf16* ga = A + (size_t)(m0 + row) * K + k0 + h2;
        const bf16* gw = W + (size_t)(n0 + row) * K + k0 + h2;
        async_b128(ga,     &sA[st][row * 32 + h2]);
        async_b128(ga + 8, &sA[st][row * 32 + h2 + 8]);
        async_b128(gw,     &sW[st][row * 32 + h2]);
        async_b128(gw + 8, &sW[st][row * 32 + h2 + 8]);
#else
        const int row = tid >> 1, h2 = (tid & 1) * 16;
        *(v16bf*)&sA[st][row * 32 + h2] = *(const v16bf*)(A + (size_t)(m0 + row) * K + k0 + h2);
        *(v16bf*)&sW[st][row * 32 + h2] = *(const v16bf*)(W + (size_t)(n0 + row) * K + k0 + h2);
#endif
    };
    auto fence = [&]() {
#if defined(USE_TDM)
        if (wave == 0) __builtin_amdgcn_s_wait_tensorcnt(0);
#elif defined(USE_ASYNC)
        wait_async0();
#endif
        __syncthreads();
    };

    const int nsteps = K >> 5;
    issue(0, 0);
    fence();
    for (int s = 0; s < nsteps; ++s) {
        const int cur = s & 1;
        if (s + 1 < nsteps) issue(cur ^ 1, (s + 1) << 5);  // overlaps with WMMAs below

        v16bf aF[4], bF[2];
        #pragma unroll
        for (int i = 0; i < 4; ++i) {
            const int r = wm + i * 16 + r16;
            aF[i] = *(const v16bf*)&sA[cur][r * 32 + akb];
        }
        #pragma unroll
        for (int j = 0; j < 2; ++j)
            bF[j] = *(const v16bf*)&sW[cur][(wn + j * 16 + r16) * 32 + bkb];

        #pragma unroll
        for (int i = 0; i < 4; ++i)
            #pragma unroll
            for (int j = 0; j < 2; ++j)
                acc[i][j] = WMMA_BF16(aF[i], bF[j], acc[i][j]);
        fence();
    }

    // epilogue
    #pragma unroll
    for (int i = 0; i < 4; ++i) {
        #pragma unroll
        for (int j = 0; j < 2; ++j) {
            #pragma unroll
            for (int v = 0; v < 8; ++v) {
                const int row = m0 + wm + i * 16 + v + 8 * hi;
                const int col = n0 + wn + j * 16 + r16;
                float val = acc[i][j][v];
                if constexpr (EPI == 0) {
                    OB[(size_t)row * N + col] = (bf16)val;
                } else if constexpr (EPI == 1) {
                    val = 0.5f * val * (1.f + erff(val * 0.7071067811865476f));
                    OB[(size_t)row * N + col] = (bf16)val;
                } else if constexpr (EPI == 2) {
                    OF[(size_t)row * N + col] = val + R[(size_t)row * N + col];
                } else if constexpr (EPI == 3) {
                    OF[(size_t)row * N + col] += val;
                } else {  // EPI == 4: per-batch transposed bf16 store [b][n][s]
                    const int b = row / Sdim, s2 = row - b * Sdim;
                    OB[((size_t)b * N + col) * Sdim + s2] = (bf16)val;
                }
            }
        }
    }
}

// ---------------------------------------------------------------------------
// Flash-style causal attention, one wave per 16-row Q tile, dk = 64.
// Q,K: [b][s][D] bf16, Vt: [b][D][s] bf16, O: [b][s][D] bf16.
// ---------------------------------------------------------------------------
__global__ __launch_bounds__(32)
void attn_kernel(const bf16* __restrict__ Q, const bf16* __restrict__ Kmat,
                 const bf16* __restrict__ Vt, bf16* __restrict__ Ob,
                 int S, int Dm, int H) {
    __shared__ bf16 sP[16 * 32];
    const int lane = threadIdx.x & 31;
    const int r16  = lane & 15;
    const int hi   = lane >> 4;
    const int qb   = blockIdx.x * 16;
    const int bh   = blockIdx.y;
    const int b    = bh / H, h = bh % H;
    const int dk   = 64;
    const size_t baseQK = ((size_t)b * S) * Dm + (size_t)h * dk;

    v16bf aQ[2];
    #pragma unroll
    for (int ks = 0; ks < 2; ++ks) {
        const bf16* p = Q + baseQK + (size_t)(qb + r16) * Dm + ks * 32 + hi * 16;
        aQ[ks] = *(const v16bf*)p;
    }

    float m[8], l[8];
    v8f O[4] = {};
    #pragma unroll
    for (int v = 0; v < 8; ++v) { m[v] = -1e30f; l[v] = 0.f; }

    const int kend = qb + 16;
    for (int j0 = 0; j0 < kend; j0 += 32) {
        v8f sc[2];
        #pragma unroll
        for (int sub = 0; sub < 2; ++sub) {
            v16bf bK[2];
            #pragma unroll
            for (int ks = 0; ks < 2; ++ks)
                bK[ks] = *(const v16bf*)(Kmat + baseQK +
                         (size_t)(j0 + sub * 16 + r16) * Dm + ks * 32 + hi * 16);
            v8f z = {};
            z = WMMA_BF16(aQ[0], bK[0], z);
            z = WMMA_BF16(aQ[1], bK[1], z);
            sc[sub] = z;
        }
        #pragma unroll
        for (int sub = 0; sub < 2; ++sub)
            #pragma unroll
            for (int v = 0; v < 8; ++v) {
                float s = sc[sub][v] * 0.125f;
                const int j    = j0 + sub * 16 + r16;
                const int qrow = qb + v + 8 * hi;
                sc[sub][v] = (j > qrow) ? -1e30f : s;
            }
        float rmax[8];
        #pragma unroll
        for (int v = 0; v < 8; ++v) rmax[v] = fmaxf(sc[0][v], sc[1][v]);
        #pragma unroll
        for (int mk = 1; mk < 16; mk <<= 1)
            #pragma unroll
            for (int v = 0; v < 8; ++v) rmax[v] = fmaxf(rmax[v], __shfl_xor(rmax[v], mk, 32));
        float sca[8], rs[8];
        #pragma unroll
        for (int v = 0; v < 8; ++v) {
            const float mn = fmaxf(m[v], rmax[v]);
            sca[v] = __expf(m[v] - mn);
            m[v]   = mn;
            const float p0 = __expf(sc[0][v] - mn);
            const float p1 = __expf(sc[1][v] - mn);
            sc[0][v] = p0; sc[1][v] = p1;
            rs[v] = p0 + p1;
        }
        #pragma unroll
        for (int mk = 1; mk < 16; mk <<= 1)
            #pragma unroll
            for (int v = 0; v < 8; ++v) rs[v] += __shfl_xor(rs[v], mk, 32);
        #pragma unroll
        for (int v = 0; v < 8; ++v) l[v] = l[v] * sca[v] + rs[v];
        #pragma unroll
        for (int g = 0; g < 4; ++g)
            #pragma unroll
            for (int v = 0; v < 8; ++v) O[g][v] *= sca[v];

        #pragma unroll
        for (int v = 0; v < 8; ++v) {
            const int row = v + 8 * hi;
            sP[row * 32 + r16]      = (bf16)sc[0][v];
            sP[row * 32 + 16 + r16] = (bf16)sc[1][v];
        }
        asm volatile("s_wait_dscnt 0" ::: "memory");
        v16bf aP = *(const v16bf*)&sP[r16 * 32 + hi * 16];

        #pragma unroll
        for (int g = 0; g < 4; ++g) {
            const bf16* p = Vt + ((size_t)b * Dm + h * dk + g * 16 + r16) * S + j0 + hi * 16;
            v16bf bV = *(const v16bf*)p;
            O[g] = WMMA_BF16(aP, bV, O[g]);
        }
        asm volatile("s_wait_dscnt 0" ::: "memory");
    }

    float invl[8];
    #pragma unroll
    for (int v = 0; v < 8; ++v) invl[v] = 1.f / l[v];
    #pragma unroll
    for (int g = 0; g < 4; ++g)
        #pragma unroll
        for (int v = 0; v < 8; ++v)
            Ob[baseQK + (size_t)(qb + v + 8 * hi) * Dm + g * 16 + r16] =
                (bf16)(O[g][v] * invl[v]);
}

// ---------------------------------------------------------------------------
// Host-side launch sequence
// ---------------------------------------------------------------------------
extern "C" void kernel_launch(void* const* d_in, const int* in_sizes, int n_in,
                              void* d_out, int out_size, void* d_ws, size_t ws_size,
                              hipStream_t stream) {
    (void)in_sizes; (void)n_in; (void)out_size; (void)ws_size;
    const int B = 2, S = 2048, D = 1024, H = 16, DFF = 4096;
    const int M = B * S;

    const float* x  = (const float*)d_in[0];
    const float* g1 = (const float*)d_in[1];
    const float* g2 = (const float*)d_in[2];
    const float* wq = (const float*)d_in[3];
    const float* wk = (const float*)d_in[4];
    const float* wv = (const float*)d_in[5];
    const float* wo = (const float*)d_in[6];
    const float* w1 = (const float*)d_in[7];
    const float* w2 = (const float*)d_in[8];
    float* out = (float*)d_out;

    bf16* p   = (bf16*)d_ws;
    bf16* wqb = p;              p += (size_t)D * D;
    bf16* wkb = p;              p += (size_t)D * D;
    bf16* wvb = p;              p += (size_t)D * D;
    bf16* wob = p;              p += (size_t)D * D;
    bf16* w1b = p;              p += (size_t)DFF * D;
    bf16* w2b = p;              p += (size_t)D * DFF;
    bf16* xn  = p;              p += (size_t)M * D;
    bf16* qb  = p;              p += (size_t)M * D;
    bf16* kb  = p;              p += (size_t)M * D;
    bf16* vt  = p;              p += (size_t)M * D;   // V^T: [b][D][S]
    bf16* at  = p;              p += (size_t)M * D;
    bf16* xn2 = p;              p += (size_t)M * D;
    bf16* hb  = p;              p += (size_t)M * DFF;

    cvt_f32_bf16<<<256, 256, 0, stream>>>(wq, wqb, D * D);
    cvt_f32_bf16<<<256, 256, 0, stream>>>(wk, wkb, D * D);
    cvt_f32_bf16<<<256, 256, 0, stream>>>(wv, wvb, D * D);
    cvt_f32_bf16<<<256, 256, 0, stream>>>(wo, wob, D * D);
    cvt_f32_bf16<<<512, 256, 0, stream>>>(w1, w1b, DFF * D);
    cvt_f32_bf16<<<512, 256, 0, stream>>>(w2, w2b, D * DFF);

    rmsnorm_kernel<<<M, 256, 0, stream>>>(x, g1, xn, D);

    dim3 gProj(D / 128, M / 128);
    gemm_bf16_wmma<0><<<gProj, 256, 0, stream>>>(xn, wqb, nullptr, nullptr, qb, M, D, D, S);
    gemm_bf16_wmma<0><<<gProj, 256, 0, stream>>>(xn, wkb, nullptr, nullptr, kb, M, D, D, S);
    gemm_bf16_wmma<4><<<gProj, 256, 0, stream>>>(xn, wvb, nullptr, nullptr, vt, M, D, D, S);

    attn_kernel<<<dim3(S / 16, B * H), 32, 0, stream>>>(qb, kb, vt, at, S, D, H);

    gemm_bf16_wmma<2><<<gProj, 256, 0, stream>>>(at, wob, x, out, nullptr, M, D, D, S);

    rmsnorm_kernel<<<M, 256, 0, stream>>>(out, g2, xn2, D);

    gemm_bf16_wmma<1><<<dim3(DFF / 128, M / 128), 256, 0, stream>>>(
        xn2, w1b, nullptr, nullptr, hb, M, DFF, D, S);

    gemm_bf16_wmma<3><<<gProj, 256, 0, stream>>>(hb, w2b, nullptr, out, nullptr, M, D, DFF, S);
}
